// MultiHeadAttention_34102040330832
// MI455X (gfx1250) — compile-verified, matrix operand order checked by reading, not run on hardware
//
#include <hip/hip_runtime.h>
#include <hip/hip_bf16.h>

typedef __attribute__((ext_vector_type(16))) _Float16 v16h;
typedef __attribute__((ext_vector_type(8)))  _Float16 v8h;
typedef __attribute__((ext_vector_type(8)))  float    v8f;
typedef __attribute__((ext_vector_type(4)))  int      v4i;

#define BATCH  2
#define SEQ    2048
#define DMODEL 1024
#define NHEAD  16
#define DHEAD  64

#define AS1 __attribute__((address_space(1)))
#define AS3 __attribute__((address_space(3)))

#if __has_builtin(__builtin_amdgcn_global_load_async_to_lds_b128) && \
    __has_builtin(__builtin_amdgcn_s_wait_asynccnt)
#define HAVE_ASYNC_LDS 1
#else
#define HAVE_ASYNC_LDS 0
#endif

// Load a 16x32 f16 A- or B-fragment from a row-major tile (ld in elements,
// ld must be a multiple of 8, base 16B-aligned).
// ISA layout (CDNA5 ISA, 16-bit 16x32): lane holds row lane%16;
// half slots [0..7] -> k = 8*(lane/16)+i, [8..15] -> k = 16 + 8*(lane/16)+(i-8).
__device__ __forceinline__ v16h load_frag(const _Float16* p, int ld) {
  const int lane = threadIdx.x & 31;
  const _Float16* q = p + (size_t)(lane & 15) * ld + ((lane >> 4) << 3);
  v8h lo = *(const v8h*)(q);
  v8h hi = *(const v8h*)(q + 16);
  v16h f;
#pragma unroll
  for (int i = 0; i < 8; ++i) { f[i] = lo[i]; f[i + 8] = hi[i]; }
  return f;
}

__device__ __forceinline__ v8f wmma32(v16h a, v16h b, v8f c) {
  return __builtin_amdgcn_wmma_f32_16x16x32_f16(false, a, false, b, (short)0, c,
                                                false, false);
}

__global__ void cvt_f16(const float* __restrict__ in, _Float16* __restrict__ out, int n) {
  int i = blockIdx.x * blockDim.x + threadIdx.x;
  if (i < n) out[i] = (_Float16)in[i];
}

// ---------------- fused QKV projection -------------------------------------
// grid (BS/64, D/64, 3), block 128 (4 waves). Wave computes a 16x64 strip.
// k-loop unrolled x2 with explicit ping-pong fragment buffers: each half
// consumes one buffer while the next k-step's loads fill the other, so the
// WMMAs overlap in-flight loads with no register-copy overhead.
__global__ void __launch_bounds__(128)
qkv_kernel(const _Float16* __restrict__ xh, const _Float16* __restrict__ wqh,
           const _Float16* __restrict__ wkh, const _Float16* __restrict__ wvh,
           const float* __restrict__ bq, const float* __restrict__ bk,
           const float* __restrict__ bv, _Float16* __restrict__ Qo,
           _Float16* __restrict__ Ko, _Float16* __restrict__ Vt) {
  const int lane = threadIdx.x & 31;
  const int wave = threadIdx.x >> 5;
  const int m0 = blockIdx.x * 64 + wave * 16;
  const int n0 = blockIdx.y * 64;
  const int z = blockIdx.z;
  const _Float16* W = (z == 0) ? wqh : (z == 1) ? wkh : wvh;
  const float* bias = (z == 0) ? bq : (z == 1) ? bk : bv;

  const _Float16* Arow = xh + (size_t)m0 * DMODEL;

  v8f acc[4] = {};
  v16h a0 = load_frag(Arow, DMODEL);
  v16h b0[4];
#pragma unroll
  for (int t = 0; t < 4; ++t)
    b0[t] = load_frag(W + (size_t)(n0 + 16 * t) * DMODEL, DMODEL);
  v16h a1, b1[4];

  for (int k0 = 0; k0 < DMODEL; k0 += 64) {
    // prefetch k0+32 into buffer 1 (always in range: k0 <= DMODEL-64)
    a1 = load_frag(Arow + k0 + 32, DMODEL);
#pragma unroll
    for (int t = 0; t < 4; ++t)
      b1[t] = load_frag(W + (size_t)(n0 + 16 * t) * DMODEL + k0 + 32, DMODEL);
#pragma unroll
    for (int t = 0; t < 4; ++t) acc[t] = wmma32(a0, b0[t], acc[t]);
    // prefetch k0+64 into buffer 0
    if (k0 + 64 < DMODEL) {
      a0 = load_frag(Arow + k0 + 64, DMODEL);
#pragma unroll
      for (int t = 0; t < 4; ++t)
        b0[t] = load_frag(W + (size_t)(n0 + 16 * t) * DMODEL + k0 + 64, DMODEL);
    }
#pragma unroll
    for (int t = 0; t < 4; ++t) acc[t] = wmma32(a1, b1[t], acc[t]);
  }
#pragma unroll
  for (int t = 0; t < 4; ++t)
#pragma unroll
    for (int r = 0; r < 8; ++r) {
      const int R = m0 + ((lane >> 4) << 3) + r;        // global row in (B*S)
      const int n = n0 + 16 * t + (lane & 15);          // col in D
      const int b = R >> 11, s = R & (SEQ - 1);
      const int h = n >> 6, dh = n & (DHEAD - 1);
      const _Float16 hv = (_Float16)(acc[t][r] + bias[n]);
      if (z == 2)       // V stored transposed: [B,H,DH,S]
        Vt[((size_t)(b * NHEAD + h) * DHEAD + dh) * SEQ + s] = hv;
      else if (z == 1)
        Ko[((size_t)(b * NHEAD + h) * SEQ + s) * DHEAD + dh] = hv;
      else
        Qo[((size_t)(b * NHEAD + h) * SEQ + s) * DHEAD + dh] = hv;
    }
}

// ---------------- flash attention ------------------------------------------
// grid (S/64, B*H), block 128. Each wave owns 16 q-rows; kv blocks of 64.
// All 4 waves process kv blocks 0..blockIdx.x (same trip count), so the
// block can cooperatively double-buffer K/V tiles in LDS via CDNA5 async
// global->LDS loads (ASYNCcnt), prefetching block kb+1 during block kb.
__global__ void __launch_bounds__(128)
attn_kernel(const _Float16* __restrict__ Q, const _Float16* __restrict__ K,
            const _Float16* __restrict__ Vt, _Float16* __restrict__ ctx) {
  const int lane = threadIdx.x & 31;
  const int wave = threadIdx.x >> 5;
  const int tid = threadIdx.x;
  const int bh = blockIdx.y;
  const int blk = blockIdx.x;
  const int q0 = blk * 64 + wave * 16;

  const _Float16* Qp = Q + (size_t)bh * SEQ * DHEAD;
  const _Float16* Kp = K + (size_t)bh * SEQ * DHEAD;
  const _Float16* Vp = Vt + (size_t)bh * DHEAD * SEQ;

  __shared__ __align__(16) _Float16 Pl[4][16][72];   // per-wave P staging
#if HAVE_ASYNC_LDS
  __shared__ __align__(16) _Float16 Ksh[2][64][64];  // [buf][key][dh]
  __shared__ __align__(16) _Float16 Vsh[2][64][64];  // [buf][dh][key]

  // Cooperative async stage of one 64x64 K tile + 64x64 V tile (8 ops/thread).
  auto stage = [&](int kv0, int buf) {
#pragma unroll
    for (int j = 0; j < 4; ++j) {
      const int c = tid + 128 * j;                 // chunk 0..511
      const int row = c >> 3, col = (c & 7) << 3;  // 8 halfs = 16B per chunk
      __builtin_amdgcn_global_load_async_to_lds_b128(
          (AS1 v4i*)(Kp + (size_t)(kv0 + row) * DHEAD + col),
          (AS3 v4i*)&Ksh[buf][row][col], 0, 0);
    }
#pragma unroll
    for (int j = 0; j < 4; ++j) {
      const int c = tid + 128 * j;
      const int row = c >> 3, col = (c & 7) << 3;
      __builtin_amdgcn_global_load_async_to_lds_b128(
          (AS1 v4i*)(Vp + (size_t)row * SEQ + kv0 + col),
          (AS3 v4i*)&Vsh[buf][row][col], 0, 0);
    }
  };
#endif

  // Q fragments held in registers for the whole kv loop (DH=64 -> 2 k-steps)
  v16h qa0 = load_frag(Qp + (size_t)q0 * DHEAD, DHEAD);
  v16h qa1 = load_frag(Qp + (size_t)q0 * DHEAD + 32, DHEAD);

  v8f o[4] = {};
  float mrun[8], lrun[8];
#pragma unroll
  for (int r = 0; r < 8; ++r) { mrun[r] = -3.0e38f; lrun[r] = 0.0f; }

  const int rowbase = q0 + ((lane >> 4) << 3);  // C-layout: row = r + 8*(lane/16)
  const float scale = 0.125f;                   // 1/sqrt(DH)

#if HAVE_ASYNC_LDS
  stage(0, 0);
#endif

  for (int kb = 0; kb <= blk; ++kb) {
    const int kv0 = kb * 64;
#if HAVE_ASYNC_LDS
    const int buf = kb & 1;
    if (kb < blk) {
      stage(kv0 + 64, buf ^ 1);
      __builtin_amdgcn_s_wait_asynccnt(8);  // prior block's 8 ops retired
    } else {
      __builtin_amdgcn_s_wait_asynccnt(0);
    }
    __syncthreads();
    const _Float16* kbase = &Ksh[buf][0][0]; const int kld = 64;
    const _Float16* vbase = &Vsh[buf][0][0]; const int vld = 64;
#else
    const _Float16* kbase = Kp + (size_t)kv0 * DHEAD; const int kld = DHEAD;
    const _Float16* vbase = Vp + kv0;                 const int vld = SEQ;
#endif
    // S = Q * K^T  (B[k][n] = K[n][k] -> same row-major fragment loader)
    v8f s[4] = {};
#pragma unroll
    for (int t = 0; t < 4; ++t) {
      v16h kb0 = load_frag(kbase + (size_t)(16 * t) * kld, kld);
      v16h kb1 = load_frag(kbase + (size_t)(16 * t) * kld + 32, kld);
      s[t] = wmma32(qa0, kb0, s[t]);
      s[t] = wmma32(qa1, kb1, s[t]);
    }
    // scale + causal mask (only the diagonal block needs masking)
    const bool diag = (kb == blk);
#pragma unroll
    for (int t = 0; t < 4; ++t)
#pragma unroll
      for (int r = 0; r < 8; ++r) {
        float v = s[t][r] * scale;
        if (diag && (kv0 + 16 * t + (lane & 15)) > (rowbase + r)) v += -1.0e9f;
        s[t][r] = v;
      }
    // row max over 64 cols: 4 per-lane values + 16-lane xor reduction
    float rmax[8];
#pragma unroll
    for (int r = 0; r < 8; ++r)
      rmax[r] = fmaxf(fmaxf(s[0][r], s[1][r]), fmaxf(s[2][r], s[3][r]));
#pragma unroll
    for (int off = 1; off < 16; off <<= 1)
#pragma unroll
      for (int r = 0; r < 8; ++r)
        rmax[r] = fmaxf(rmax[r], __shfl_xor(rmax[r], off, 32));

    float newm[8], alpha[8], rsum[8];
#pragma unroll
    for (int r = 0; r < 8; ++r) {
      newm[r] = fmaxf(mrun[r], rmax[r]);
      alpha[r] = __expf(mrun[r] - newm[r]);
    }
#pragma unroll
    for (int t = 0; t < 4; ++t)
#pragma unroll
      for (int r = 0; r < 8; ++r)
        s[t][r] = __expf(s[t][r] - newm[r]);
#pragma unroll
    for (int r = 0; r < 8; ++r)
      rsum[r] = (s[0][r] + s[1][r]) + (s[2][r] + s[3][r]);
#pragma unroll
    for (int off = 1; off < 16; off <<= 1)
#pragma unroll
      for (int r = 0; r < 8; ++r)
        rsum[r] += __shfl_xor(rsum[r], off, 32);
#pragma unroll
    for (int r = 0; r < 8; ++r) {
      lrun[r] = lrun[r] * alpha[r] + rsum[r];
      mrun[r] = newm[r];
    }
    // rescale O, stage P (f16) into LDS in row-major for A-fragment reload
#pragma unroll
    for (int t = 0; t < 4; ++t)
#pragma unroll
      for (int r = 0; r < 8; ++r) {
        o[t][r] *= alpha[r];
        Pl[wave][((lane >> 4) << 3) + r][16 * t + (lane & 15)] = (_Float16)s[t][r];
      }
    // reload P as A-fragments (per-wave LDS region, same-wave DS ordering)
    const _Float16* pb = &Pl[wave][lane & 15][(lane >> 4) << 3];
    v16h pa0, pa1;
    {
      v8h l0 = *(const v8h*)(pb);
      v8h h0 = *(const v8h*)(pb + 16);
      v8h l1 = *(const v8h*)(pb + 32);
      v8h h1 = *(const v8h*)(pb + 48);
#pragma unroll
      for (int i = 0; i < 8; ++i) {
        pa0[i] = l0[i]; pa0[i + 8] = h0[i];
        pa1[i] = l1[i]; pa1[i + 8] = h1[i];
      }
    }
    // O += P * V   (B[k][n] = Vt[n][k], contiguous in k)
#pragma unroll
    for (int t = 0; t < 4; ++t) {
      v16h vb0 = load_frag(vbase + (size_t)(16 * t) * vld, vld);
      v16h vb1 = load_frag(vbase + (size_t)(16 * t) * vld + 32, vld);
      o[t] = wmma32(pa0, vb0, o[t]);
      o[t] = wmma32(pa1, vb1, o[t]);
    }
#if HAVE_ASYNC_LDS
    __syncthreads();   // all waves done reading buf before it is re-staged
#endif
  }
  // epilogue: normalize, write merged-head context (B,S,D) f16
  const int b = bh >> 4, h = bh & (NHEAD - 1);
#pragma unroll
  for (int t = 0; t < 4; ++t)
#pragma unroll
    for (int r = 0; r < 8; ++r) {
      const int srow = rowbase + r;
      const int dh = 16 * t + (lane & 15);
      const float val = o[t][r] / lrun[r];
      ctx[((size_t)b * SEQ + srow) * DMODEL + h * DHEAD + dh] = (_Float16)val;
    }
}

// ---------------- output projection ----------------------------------------
__global__ void __launch_bounds__(128)
oproj_kernel(const _Float16* __restrict__ ctx, const _Float16* __restrict__ woh,
             const float* __restrict__ bo, float* __restrict__ out) {
  const int lane = threadIdx.x & 31;
  const int wave = threadIdx.x >> 5;
  const int m0 = blockIdx.x * 64 + wave * 16;
  const int n0 = blockIdx.y * 64;

  const _Float16* Arow = ctx + (size_t)m0 * DMODEL;

  v8f acc[4] = {};
  v16h a0 = load_frag(Arow, DMODEL);
  v16h b0[4];
#pragma unroll
  for (int t = 0; t < 4; ++t)
    b0[t] = load_frag(woh + (size_t)(n0 + 16 * t) * DMODEL, DMODEL);
  v16h a1, b1[4];

  for (int k0 = 0; k0 < DMODEL; k0 += 64) {
    a1 = load_frag(Arow + k0 + 32, DMODEL);
#pragma unroll
    for (int t = 0; t < 4; ++t)
      b1[t] = load_frag(woh + (size_t)(n0 + 16 * t) * DMODEL + k0 + 32, DMODEL);
#pragma unroll
    for (int t = 0; t < 4; ++t) acc[t] = wmma32(a0, b0[t], acc[t]);
    if (k0 + 64 < DMODEL) {
      a0 = load_frag(Arow + k0 + 64, DMODEL);
#pragma unroll
      for (int t = 0; t < 4; ++t)
        b0[t] = load_frag(woh + (size_t)(n0 + 16 * t) * DMODEL + k0 + 64, DMODEL);
    }
#pragma unroll
    for (int t = 0; t < 4; ++t) acc[t] = wmma32(a1, b1[t], acc[t]);
  }
#pragma unroll
  for (int t = 0; t < 4; ++t)
#pragma unroll
    for (int r = 0; r < 8; ++r) {
      const int R = m0 + ((lane >> 4) << 3) + r;
      const int n = n0 + 16 * t + (lane & 15);
      out[(size_t)R * DMODEL + n] = acc[t][r] + bo[n];
    }
}

extern "C" void kernel_launch(void* const* d_in, const int* in_sizes, int n_in,
                              void* d_out, int out_size, void* d_ws, size_t ws_size,
                              hipStream_t stream) {
  (void)in_sizes; (void)n_in; (void)out_size; (void)ws_size;
  const float* x  = (const float*)d_in[0];
  // d_in[1] = causal mask (implicit in kernel; not read)
  const float* wq = (const float*)d_in[2];
  const float* bq = (const float*)d_in[3];
  const float* wk = (const float*)d_in[4];
  const float* bk = (const float*)d_in[5];
  const float* wv = (const float*)d_in[6];
  const float* bv = (const float*)d_in[7];
  const float* wo = (const float*)d_in[8];
  const float* bo = (const float*)d_in[9];
  float* out = (float*)d_out;

  const size_t XN = (size_t)BATCH * SEQ * DMODEL;  // 4,194,304
  const size_t WN = (size_t)DMODEL * DMODEL;       // 1,048,576
  _Float16* xh  = (_Float16*)d_ws;
  _Float16* wqh = xh + XN;
  _Float16* wkh = wqh + WN;
  _Float16* wvh = wkh + WN;
  _Float16* woh = wvh + WN;
  _Float16* Qh  = woh + WN;
  _Float16* Kh  = Qh + XN;
  _Float16* Vth = Kh + XN;
  _Float16* ctx = Vth + XN;   // total ~48 MB of workspace

  cvt_f16<<<(unsigned)((XN + 255) / 256), 256, 0, stream>>>(x, xh, (int)XN);
  cvt_f16<<<(unsigned)((WN + 255) / 256), 256, 0, stream>>>(wq, wqh, (int)WN);
  cvt_f16<<<(unsigned)((WN + 255) / 256), 256, 0, stream>>>(wk, wkh, (int)WN);
  cvt_f16<<<(unsigned)((WN + 255) / 256), 256, 0, stream>>>(wv, wvh, (int)WN);
  cvt_f16<<<(unsigned)((WN + 255) / 256), 256, 0, stream>>>(wo, woh, (int)WN);

  dim3 gq(BATCH * SEQ / 64, DMODEL / 64, 3);
  qkv_kernel<<<gq, 128, 0, stream>>>(xh, wqh, wkh, wvh, bq, bk, bv, Qh, Kh, Vth);

  dim3 ga(SEQ / 64, BATCH * NHEAD);
  attn_kernel<<<ga, 128, 0, stream>>>(Qh, Kh, Vth, ctx);

  dim3 go(BATCH * SEQ / 64, DMODEL / 64);
  oproj_kernel<<<go, 128, 0, stream>>>(ctx, woh, bo, out);
}
